// TrainedPadder_73512660238964
// MI455X (gfx1250) — matrix operand order repaired and checked by reading, not compiled, not run on hardware
//
#include <hip/hip_runtime.h>

typedef _Float16 v16h __attribute__((ext_vector_type(16)));
typedef _Float16 v8h  __attribute__((ext_vector_type(8)));
typedef __fp16   fh2  __attribute__((ext_vector_type(2)));   // cvt_pkrtz result type
typedef float    v8f  __attribute__((ext_vector_type(8)));
typedef float    f2   __attribute__((ext_vector_type(2)));
typedef float    f4   __attribute__((ext_vector_type(4)));

#define NB   8
#define NC   128
#define HH   384
#define WW   384
#define OH   388
#define OW   388
#define KTOT 1152      // C * 3 * 3
#define NI_NJ 764      // ni*nj per edge (2*382 or 382*2)
#define CH_OUT 150544  // 388*388

// ---------------------------------------------------------------------------
// Kernel 1: edge convolutions as im2col GEMM on WMMA (f16 in, f32 accumulate)
// grid = (382 column-tiles, 4 edges), block = 256 (8 wave32s)
// wave w computes oc tile [16w, 16w+16) x 16 spatial positions, K = 1152
// ---------------------------------------------------------------------------
__global__ __launch_bounds__(256) void edge_conv_wmma(
    const float* __restrict__ x, float* __restrict__ out,
    const float* __restrict__ wu, const float* __restrict__ bu,
    const float* __restrict__ wd, const float* __restrict__ bd,
    const float* __restrict__ wl, const float* __restrict__ bl,
    const float* __restrict__ wr, const float* __restrict__ br)
{
  __shared__ __align__(16) _Float16 smem[16 * 1160];  // B-panel, padded rows

  const int e = blockIdx.y;
  const float* w;  const float* bias;
  int rbase, cbase, nj, orow, ocol;
  switch (e) {
    case 0:  w = wu; bias = bu; rbase = 0;   cbase = 0;   nj = 382; orow = 0;   ocol = 3;   break;
    case 1:  w = wd; bias = bd; rbase = 380; cbase = 0;   nj = 382; orow = 386; ocol = 3;   break;
    case 2:  w = wl; bias = bl; rbase = 0;   cbase = 0;   nj = 2;   orow = 3;   ocol = 0;   break;
    default: w = wr; bias = br; rbase = 0;   cbase = 380; nj = 2;   orow = 3;   ocol = 386; break;
  }

  const int tid  = threadIdx.x;
  const int tile = blockIdx.x;   // 0..381

  // ---- stage im2col B-panel: smem[nloc*1160 + k] = f16(x patch value) ----
  {
    const int nloc = tid & 15;
    const int n    = tile * 16 + nloc;       // global spatial index
    int b   = n / NI_NJ;
    int rem = n - b * NI_NJ;
    int i   = rem / nj;
    int j   = rem - i * nj;
    const int rr  = rbase + i;
    const int cc0 = cbase + j;
    const float* xb = x + (unsigned)(b * NC) * (HH * WW);
    _Float16* dst = smem + nloc * 1160;
    for (int k = (tid >> 4); k < KTOT; k += 16) {
      int ic = k / 9;
      int r9 = k - ic * 9;
      int kh = r9 / 3;
      int kw = r9 - kh * 3;
      float v = xb[(unsigned)(ic * HH + rr + kh) * WW + cc0 + kw];
      dst[k] = (_Float16)v;
    }
  }
  __syncthreads();

  // ---- per-wave 16x16 GEMM tile ----
  const int wv   = tid >> 5;   // wave id 0..7  -> oc tile
  const int lane = tid & 31;
  const int m    = lane & 15;  // A row (oc) / B col (n) within tile
  const int hi   = lane >> 4;
  const int ocb  = wv * 16;

  v8f acc;
  #pragma unroll
  for (int v = 0; v < 8; ++v) acc[v] = bias[ocb + hi * 8 + v];   // D row = hi*8+v

  const float*    wrow = w + (unsigned)(ocb + m) * KTOT;         // weight row, K-contig
  const _Float16* bp   = smem + m * 1160;                        // B column, K-contig

  for (int kb = 0; kb < 36; ++kb) {
    const int k0 = kb * 32;
    // A operand: lane holds K-chunks [k0+hi*8, +8) and [k0+16+hi*8, +8)
    const f4* wp = (const f4*)(wrow + k0 + hi * 8);
    f4 w0 = wp[0];
    f4 w1 = wp[1];
    const f4* wq = (const f4*)(wrow + k0 + 16 + hi * 8);
    f4 w2 = wq[0];
    f4 w3 = wq[1];
    // packed f32->f16 converts: one v_cvt_pk per A-operand VGPR
    fh2 p0 = __builtin_amdgcn_cvt_pkrtz(w0[0], w0[1]);
    fh2 p1 = __builtin_amdgcn_cvt_pkrtz(w0[2], w0[3]);
    fh2 p2 = __builtin_amdgcn_cvt_pkrtz(w1[0], w1[1]);
    fh2 p3 = __builtin_amdgcn_cvt_pkrtz(w1[2], w1[3]);
    fh2 p4 = __builtin_amdgcn_cvt_pkrtz(w2[0], w2[1]);
    fh2 p5 = __builtin_amdgcn_cvt_pkrtz(w2[2], w2[3]);
    fh2 p6 = __builtin_amdgcn_cvt_pkrtz(w3[0], w3[1]);
    fh2 p7 = __builtin_amdgcn_cvt_pkrtz(w3[2], w3[3]);
    v16h a;
    a[0]  = (_Float16)p0[0]; a[1]  = (_Float16)p0[1];
    a[2]  = (_Float16)p1[0]; a[3]  = (_Float16)p1[1];
    a[4]  = (_Float16)p2[0]; a[5]  = (_Float16)p2[1];
    a[6]  = (_Float16)p3[0]; a[7]  = (_Float16)p3[1];
    a[8]  = (_Float16)p4[0]; a[9]  = (_Float16)p4[1];
    a[10] = (_Float16)p5[0]; a[11] = (_Float16)p5[1];
    a[12] = (_Float16)p6[0]; a[13] = (_Float16)p6[1];
    a[14] = (_Float16)p7[0]; a[15] = (_Float16)p7[1];
    // B operand: lane holds K-chunks [k0+hi*8, +8) and [k0+16+hi*8, +8) of its column
    v8h b0 = *(const v8h*)(bp + k0 + hi * 8);
    v8h b1 = *(const v8h*)(bp + k0 + 16 + hi * 8);
    v16h bm;
    #pragma unroll
    for (int t = 0; t < 8; ++t) { bm[t] = b0[t]; bm[8 + t] = b1[t]; }

    acc = __builtin_amdgcn_wmma_f32_16x16x32_f16(
        /*neg_a=*/false, a, /*neg_b=*/false, bm,
        /*c_mod=*/(short)0, acc, /*reuse_a=*/false, /*reuse_b=*/false);
  }

  // ---- scatter 16x16 tile into the padded output border ----
  {
    const int n = tile * 16 + m;
    int b   = n / NI_NJ;
    int rem = n - b * NI_NJ;
    int i   = rem / nj;
    int j   = rem - i * nj;
    float* o = out + (unsigned)((b * NC + ocb + hi * 8) * OH + orow + i) * OW + ocol + j;
    #pragma unroll
    for (int v = 0; v < 8; ++v) o[(unsigned)v * CH_OUT] = acc[v];
  }
}

// ---------------------------------------------------------------------------
// Kernel 2: corner convolutions (tiny: 4 x [8,128,3x3] outputs, VALU fp32)
// flat 36864 threads; L-shaped scatter skips the one interior-covered cell
// ---------------------------------------------------------------------------
__global__ __launch_bounds__(256) void corner_conv(
    const float* __restrict__ x, float* __restrict__ out,
    const float* __restrict__ w0, const float* __restrict__ c0,
    const float* __restrict__ w1, const float* __restrict__ c1,
    const float* __restrict__ w2, const float* __restrict__ c2,
    const float* __restrict__ w3, const float* __restrict__ c3)
{
  int flat = blockIdx.x * 256 + threadIdx.x;   // 0..36863
  int cell = flat % 9;
  int t    = flat / 9;
  int oc   = t % NC;
  int t2   = t / NC;
  int b    = t2 % NB;
  int corner = t2 / NB;                        // 0=ul 1=ur 2=bl 3=br
  int i = cell / 3, j = cell % 3;

  const float* w; const float* bias;
  switch (corner) {
    case 0:  w = w0; bias = c0; break;
    case 1:  w = w1; bias = c1; break;
    case 2:  w = w2; bias = c2; break;
    default: w = w3; bias = c3; break;
  }
  const int xr0 = (corner >= 2) ? 379 : 0;
  const int xc0 = (corner & 1) ? 379 : 0;
  const int or0 = (corner >= 2) ? 385 : 0;
  const int oc0 = (corner & 1) ? 385 : 0;
  const int si  = (corner >= 2) ? 0 : 2;   // skipped cell (covered by interior)
  const int sj  = (corner & 1) ? 0 : 2;
  if (i == si && j == sj) return;

  const float* xb = x + (unsigned)((b * NC) * HH + xr0 + i) * WW + xc0 + j;
  const float* wr = w + (unsigned)oc * KTOT;
  float s = bias[oc];
  for (int ic = 0; ic < NC; ++ic) {
    const float* xp = xb + (unsigned)ic * (HH * WW);
    const float* wp = wr + ic * 9;
    #pragma unroll
    for (int kh = 0; kh < 3; ++kh)
      #pragma unroll
      for (int kw = 0; kw < 3; ++kw)
        s = fmaf(xp[kh * WW + kw], wp[kh * 3 + kw], s);
  }
  out[(unsigned)((b * NC + oc) * OH + or0 + i) * OW + oc0 + j] = s;
}

// ---------------------------------------------------------------------------
// Kernel 3: interior pad-copy, nontemporal float2 streams (the 1.2 GB mover)
// 75,497,472 float2 elements; block of 256 handles 768 (3 strided each)
// ---------------------------------------------------------------------------
__global__ __launch_bounds__(256) void pad_copy(const float* __restrict__ x,
                                                float* __restrict__ out)
{
  const unsigned base = blockIdx.x * 768u + threadIdx.x;
  #pragma unroll
  for (int s = 0; s < 3; ++s) {
    unsigned e   = base + (unsigned)s * 256u;
    unsigned row = e / 192u;             // 0..393215 = (b*C)*384 + y
    unsigned c2  = e - row * 192u;       // float2 column 0..191
    unsigned bc  = row / 384u;
    unsigned y   = row - bc * 384u;
    const f2* src = (const f2*)x + (bc * 384u + y) * 192u + c2;
    float* dstf = out + (unsigned)bc * CH_OUT + (y + 2u) * OW + 2u + 2u * c2;
    f2 v = __builtin_nontemporal_load(src);
    __builtin_nontemporal_store(v, (f2*)dstf);
  }
}

// ---------------------------------------------------------------------------
extern "C" void kernel_launch(void* const* d_in, const int* in_sizes, int n_in,
                              void* d_out, int out_size, void* d_ws, size_t ws_size,
                              hipStream_t stream)
{
  (void)in_sizes; (void)n_in; (void)out_size; (void)d_ws; (void)ws_size;
  const float* x    = (const float*)d_in[0];
  const float* w_u  = (const float*)d_in[1];
  const float* b_u  = (const float*)d_in[2];
  const float* w_d  = (const float*)d_in[3];
  const float* b_d  = (const float*)d_in[4];
  const float* w_l  = (const float*)d_in[5];
  const float* b_l  = (const float*)d_in[6];
  const float* w_r  = (const float*)d_in[7];
  const float* b_r  = (const float*)d_in[8];
  const float* w_ul = (const float*)d_in[9];
  const float* b_ul = (const float*)d_in[10];
  const float* w_ur = (const float*)d_in[11];
  const float* b_ur = (const float*)d_in[12];
  const float* w_bl = (const float*)d_in[13];
  const float* b_bl = (const float*)d_in[14];
  const float* w_br = (const float*)d_in[15];
  const float* b_br = (const float*)d_in[16];
  float* out = (float*)d_out;

  // edges: 382 column-tiles x 4 edges, 8 wave32s per block
  edge_conv_wmma<<<dim3(382, 4), 256, 0, stream>>>(
      x, out, w_u, b_u, w_d, b_d, w_l, b_l, w_r, b_r);
  // corners: 36864 threads
  corner_conv<<<144, 256, 0, stream>>>(
      x, out, w_ul, b_ul, w_ur, b_ur, w_bl, b_bl, w_br, b_br);
  // interior copy last (nontemporal stream, keeps weights L2-hot for convs)
  pad_copy<<<98304, 256, 0, stream>>>(x, out);
}
